// Decoder_48404281426081
// MI455X (gfx1250) — matrix operand order repaired and checked by reading, not compiled, not run on hardware
//
#include <hip/hip_runtime.h>
#include <hip/hip_bf16.h>

typedef _Float16 half_t;
typedef __attribute__((ext_vector_type(16))) _Float16 v16h;
typedef __attribute__((ext_vector_type(8)))  _Float16 v8h;
typedef __attribute__((ext_vector_type(4)))  _Float16 v4h;
typedef __attribute__((ext_vector_type(8)))  float    v8f;
typedef __attribute__((ext_vector_type(4)))  unsigned int u32x4;
typedef __attribute__((ext_vector_type(8)))  int i32x8;
typedef __attribute__((ext_vector_type(4)))  int i32x4;

#define KP 96      // padded K (3 chunks of 32)
#define HP 80      // padded hidden width (5 tiles of 16)
#define MR 128     // rows per block (2 x 16-row tiles per wave)
#define TPB 128    // 4 waves of 32

#define WBYTES (2 * HP * KP * 2 + 16 * KP * 2 + 176 * 4)  // 33792 + 704 = 34496 B
#define WB_DWORDS (WBYTES / 4)                            // 8624

#if defined(__gfx1250__) && __has_builtin(__builtin_amdgcn_tensor_load_to_lds)
#define USE_TDM 1
#endif

// ---------- monotonic float<->uint key mapping for radix select ----------
__device__ __forceinline__ unsigned f2key(float f) {
  unsigned u = __float_as_uint(f);
  return (u & 0x80000000u) ? ~u : (u | 0x80000000u);
}
__device__ __forceinline__ float key2f(unsigned k) {
  unsigned u = (k & 0x80000000u) ? (k & 0x7FFFFFFFu) : ~k;
  return __uint_as_float(u);
}

// ---------- group-of-8 local max + masked sort keys ----------
__global__ void k_localmax(const float* __restrict__ pred, unsigned* __restrict__ keys,
                           unsigned char* __restrict__ flags, int ngroups) {
  int g = blockIdx.x * blockDim.x + threadIdx.x;
  if (g >= ngroups) return;
  int base = g * 8;
  float v[8];
#pragma unroll
  for (int j = 0; j < 8; j++) v[j] = pred[base + j];
  int idx = 0; float best = v[0];
#pragma unroll
  for (int j = 1; j < 8; j++) if (v[j] > best) { best = v[j]; idx = j; }  // first-max, like argmax
#pragma unroll
  for (int j = 0; j < 8; j++) {
    flags[base + j] = (unsigned char)(j == idx);
    float m = (j == idx) ? __builtin_inff() : v[j];
    keys[base + j] = f2key(m);
  }
}

__global__ void k_zero_w(unsigned* p, int nwords) {
  for (int i = blockIdx.x * blockDim.x + threadIdx.x; i < nwords; i += gridDim.x * blockDim.x)
    p[i] = 0u;
}

// state[0..1]=prefix, state[2..3]=rank(1-indexed), state[4..5]=thr bits
__global__ void k_sel_init(unsigned* hist, unsigned* state, const int* pnum, int n_per) {
  int t = threadIdx.x;
  hist[t] = 0u; hist[t + 256] = 0u;
  if (t < 2) { state[t] = 0u; state[2 + t] = (unsigned)(n_per - pnum[0]); }
}

__global__ void k_hist(const unsigned* __restrict__ keys, const unsigned* __restrict__ state,
                       unsigned* __restrict__ hist, int N, int n_per, int pass) {
  __shared__ unsigned lh[512];
  for (int i = threadIdx.x; i < 512; i += blockDim.x) lh[i] = 0u;
  __syncthreads();
  int shift = 24 - 8 * pass;
  unsigned p0 = state[0], p1 = state[1];
  for (int i = blockIdx.x * blockDim.x + threadIdx.x; i < N; i += gridDim.x * blockDim.x) {
    int b = (i >= n_per) ? 1 : 0;
    unsigned key = keys[i];
    bool match = true;
    if (pass > 0) match = ((key >> (shift + 8)) == (b ? p1 : p0));
    if (match) atomicAdd(&lh[(b << 8) + ((key >> shift) & 255u)], 1u);
  }
  __syncthreads();
  for (int i = threadIdx.x; i < 512; i += blockDim.x) {
    unsigned c = lh[i];
    if (c) atomicAdd(&hist[i], c);
  }
}

__global__ void k_select(unsigned* hist, unsigned* state, int pass) {
  int t = threadIdx.x;
  if (t >= 2) return;
  unsigned kR = state[2 + t];
  unsigned cum = 0, before = 0;
  int bsel = -1;
  for (int j = 0; j < 256; j++) {
    unsigned h = hist[t * 256 + j];
    if (bsel < 0 && cum + h >= kR) { bsel = j; before = cum; }
    cum += h;
    hist[t * 256 + j] = 0u;                 // clear for next pass
  }
  if (bsel < 0) bsel = 255;
  state[t] = (state[t] << 8) | (unsigned)bsel;
  state[2 + t] = kR - before;
  if (pass == 3) ((float*)state)[4 + t] = key2f(state[t]);
}

__global__ void k_keep(const float* __restrict__ pred, const unsigned char* __restrict__ flags,
                       const unsigned* __restrict__ state, unsigned char* __restrict__ keep,
                       int N, int n_per) {
  int i = blockIdx.x * blockDim.x + threadIdx.x;
  if (i >= N) return;
  int b = (i >= n_per) ? 1 : 0;
  float thr = ((const float*)state)[4 + b];
  keep[i] = (unsigned char)((pred[i] > thr) || flags[i]);
}

__global__ void k_scatter(const int* __restrict__ idx, unsigned char* __restrict__ ktar, int M) {
  int i = blockIdx.x * blockDim.x + threadIdx.x;
  if (i < M) ktar[idx[i]] = 1;
}

// ---------- BCE-with-logits, deterministic 2-stage reduction ----------
__global__ void k_bce_partial(const float* __restrict__ pred, const unsigned char* __restrict__ ktar,
                              float* __restrict__ partial, int N) {
  __shared__ float s[256];
  float acc = 0.f;
  for (int i = blockIdx.x * blockDim.x + threadIdx.x; i < N; i += gridDim.x * blockDim.x) {
    float x = pred[i];
    float t = ktar[i] ? 1.f : 0.f;
    acc += fmaxf(x, 0.f) - x * t + log1pf(expf(-fabsf(x)));
  }
  s[threadIdx.x] = acc; __syncthreads();
  for (int o = 128; o > 0; o >>= 1) { if (threadIdx.x < o) s[threadIdx.x] += s[threadIdx.x + o]; __syncthreads(); }
  if (threadIdx.x == 0) partial[blockIdx.x] = s[0];
}

__global__ void k_bce_final(const float* __restrict__ partial, float* __restrict__ out, int nblk, int N) {
  __shared__ float s[256];
  float acc = 0.f;
  for (int i = threadIdx.x; i < nblk; i += 256) acc += partial[i];
  s[threadIdx.x] = acc; __syncthreads();
  for (int o = 128; o > 0; o >>= 1) { if (threadIdx.x < o) s[threadIdx.x] += s[threadIdx.x + o]; __syncthreads(); }
  if (threadIdx.x == 0) out[0] = s[0] / (float)N;
}

// ---------- weight prep: f32 -> f16, transposed to [n][k], zero-padded ----------
// wbuf layout (halves): W1t[HP][KP] | W2t[HP][KP] | W3t[16][KP];  bbuf (floats, contiguous after): b1[80] b2[80] b3[16]
__global__ void k_prep(const float* W1, const float* b1, const float* W2, const float* b2,
                       const float* W3, const float* b3, half_t* wbuf, float* bbuf) {
  const int total = HP * KP * 2 + 16 * KP;   // 16896
  for (int i = blockIdx.x * blockDim.x + threadIdx.x; i < total; i += gridDim.x * blockDim.x) {
    float v = 0.f;
    if (i < HP * KP)            { int n = i / KP, k = i % KP; if (n < 66 && k < 66) v = W1[k * 66 + n]; }
    else if (i < 2 * HP * KP)   { int j = i - HP * KP; int n = j / KP, k = j % KP; if (n < 66 && k < 66) v = W2[k * 66 + n]; }
    else                        { int j = i - 2 * HP * KP; int n = j / KP, k = j % KP; if (n < 3 && k < 66) v = W3[k * 3 + n]; }
    wbuf[i] = (half_t)v;
  }
  int t = blockIdx.x * blockDim.x + threadIdx.x;
  if (t < 176) {
    float v = 0.f;
    if (t < 80)       { if (t < 66)       v = b1[t]; }
    else if (t < 160) { int j = t - 80;  if (j < 66) v = b2[j]; }
    else              { int j = t - 160; if (j < 3)  v = b3[j]; }
    bbuf[t] = v;
  }
}

// ---------- fused MLP via v_wmma_f32_16x16x32_f16 + TDM weight DMA ----------
union Frag { v16h v; v8h h[2]; };

__global__ __launch_bounds__(TPB) void k_mlp(const float* __restrict__ fea,
                                             const unsigned char* __restrict__ keep,
                                             const half_t* __restrict__ wbuf,
                                             float* __restrict__ out, int N) {
  __shared__ alignas(16) half_t A[MR][KP];            // 24576 B activations (in-place across layers)
  __shared__ alignas(16) unsigned char blob[WBYTES];  // 34496 B: W1t|W2t|W3t halves, then bias floats
  half_t* W  = (half_t*)blob;
  float*  Bv = (float*)(blob + 2 * HP * KP * 2 + 16 * KP * 2);

  const int t = threadIdx.x;
  const int rowBase = blockIdx.x * MR;

#ifdef USE_TDM
  // One TDM DMA for the whole weight+bias blob: 1-D tile, 4-byte elements, 8624 dwords.
  if (t < 32) {
    unsigned long long ga = (unsigned long long)(const void*)wbuf;
    unsigned lds_off = (unsigned)(unsigned long long)(const void*)blob;  // low 32 bits = LDS byte offset
    u32x4 g0;
    g0[0] = 1u;                                              // count=1, gather off, user mode
    g0[1] = lds_off;                                         // lds_addr
    g0[2] = (unsigned)(ga & 0xFFFFFFFFull);                  // global_addr[31:0]
    g0[3] = (unsigned)((ga >> 32) & 0x01FFFFFFull) | (2u << 30);  // global_addr[56:32] | type=2
    i32x8 g1;
    g1[0] = 2 << 16;                                         // workgroup_mask=0, data_size=4B
    g1[1] = (int)((WB_DWORDS & 0xFFFF) << 16);               // tensor_dim0[15:0]
    g1[2] = (int)(((WB_DWORDS >> 16) & 0xFFFF) | (1 << 16)); // tensor_dim0[31:16] | tensor_dim1=1
    g1[3] = (int)((WB_DWORDS & 0xFFFF) << 16);               // tile_dim0
    g1[4] = 1;                                               // tile_dim1=1, tile_dim2=0
    g1[5] = WB_DWORDS;                                       // tensor_dim0_stride[31:0]
    g1[6] = 0; g1[7] = 0;
    i32x4 gz = {0, 0, 0, 0};
#if __clang_major__ >= 23
    i32x8 gz8 = {0, 0, 0, 0, 0, 0, 0, 0};
    __builtin_amdgcn_tensor_load_to_lds(g0, g1, gz, gz, gz8, 0);
#else
    __builtin_amdgcn_tensor_load_to_lds(g0, g1, gz, gz, 0);
#endif
  }
#else
  {
    uint4* dst = (uint4*)blob;
    const uint4* src = (const uint4*)wbuf;
    for (int i = t; i < WBYTES / 16; i += TPB) dst[i] = src[i];
  }
#endif

  // cooperative A load (overlaps the weight DMA): 128 rows x 16 float4 of fea, f32->f16
  {
    const float4* f4 = (const float4*)fea;
    for (int i = t; i < MR * 16; i += TPB) {
      int r = i >> 4, c = i & 15;
      float4 v = f4[(size_t)(rowBase + r) * 16 + c];
      v4h h; h[0] = (half_t)v.x; h[1] = (half_t)v.y; h[2] = (half_t)v.z; h[3] = (half_t)v.w;
      *(v4h*)&A[r][c * 4] = h;
    }
    for (int r = t; r < MR; r += TPB) {
      unsigned char kf = keep[rowBase + r];
      A[r][64] = kf ? (half_t)1.0f : (half_t)0.0f;
      A[r][65] = (half_t)1.0f;
#pragma unroll
      for (int c = 66; c < KP; c++) A[r][c] = (half_t)0.0f;  // pad stays 0 through all layers
    }
  }
#ifdef USE_TDM
  if (t < 32) __builtin_amdgcn_s_wait_tensorcnt(0);
#endif
  __syncthreads();

  const int lane = t & 31;
  const int wv   = t >> 5;       // wave id 0..3 -> 32-row strip (two 16-row M-tiles)
  const int mi   = lane & 15;
  const int hh   = lane >> 4;

#pragma unroll 1
  for (int sub = 0; sub < 2; sub++) {
    const int r0 = wv * 32 + sub * 16;

    // ---- layers 1 & 2: 66x66, 5 n-tiles x 3 k-chunks, bias+relu, write back in place ----
#pragma unroll 1
    for (int layer = 0; layer < 2; layer++) {
      const half_t* Wt  = &W[layer * HP * KP];
      const float* bias = &Bv[layer * 80];
      Frag a[3];
#pragma unroll
      for (int kc = 0; kc < 3; kc++) {     // A frag: lanes 0-15 K=kb+0..7/16..23, lanes 16-31 +8
        int kb = kc * 32;
        a[kc].h[0] = *(const v8h*)&A[r0 + mi][kb + 8 * hh];
        a[kc].h[1] = *(const v8h*)&A[r0 + mi][kb + 16 + 8 * hh];
      }
      v8f acc[5];
#pragma unroll
      for (int nt = 0; nt < 5; nt++) {
        float bz = bias[nt * 16 + mi];     // C/D: lane = column -> bias is per-lane constant
        v8f c;
#pragma unroll
        for (int q = 0; q < 8; q++) c[q] = bz;
#pragma unroll
        for (int kc = 0; kc < 3; kc++) {   // B frag: lane = column, K=kb+8hh+0..7 / kb+16+8hh+0..7
          Frag b;
          const half_t* bp = &Wt[(nt * 16 + mi) * KP + kc * 32 + 8 * hh];
          b.h[0] = *(const v8h*)bp;
          b.h[1] = *(const v8h*)(bp + 16);
          c = __builtin_amdgcn_wmma_f32_16x16x32_f16(false, a[kc].v, false, b.v,
                                                     (short)0, c, false, false);
        }
        acc[nt] = c;
      }
      // relu + write h back to this wave's own row strip (no cross-wave sharing)
#pragma unroll
      for (int nt = 0; nt < 5; nt++) {
#pragma unroll
        for (int q = 0; q < 8; q++) {
          float hv = acc[nt][q];
          hv = hv > 0.f ? hv : 0.f;
          A[r0 + q + 8 * hh][nt * 16 + mi] = (half_t)hv;
        }
      }
    }

    // ---- layer 3: 66x3 (one 16-wide tile), clip*255, keep-gate, store ----
    {
      const half_t* Wt = &W[2 * HP * KP];
      Frag a[3];
#pragma unroll
      for (int kc = 0; kc < 3; kc++) {
        int kb = kc * 32;
        a[kc].h[0] = *(const v8h*)&A[r0 + mi][kb + 8 * hh];
        a[kc].h[1] = *(const v8h*)&A[r0 + mi][kb + 16 + 8 * hh];
      }
      float bz = Bv[160 + mi];
      v8f c;
#pragma unroll
      for (int q = 0; q < 8; q++) c[q] = bz;
#pragma unroll
      for (int kc = 0; kc < 3; kc++) {
        Frag b;
        const half_t* bp = &Wt[mi * KP + kc * 32 + 8 * hh];
        b.h[0] = *(const v8h*)bp;
        b.h[1] = *(const v8h*)(bp + 16);
        c = __builtin_amdgcn_wmma_f32_16x16x32_f16(false, a[kc].v, false, b.v,
                                                   (short)0, c, false, false);
      }
#pragma unroll
      for (int q = 0; q < 8; q++) {
        int grow = rowBase + r0 + q + 8 * hh;
        if (mi < 3) {
          unsigned char kf = keep[grow];
          float v = c[q];
          v = v < 0.f ? 0.f : (v > 1.f ? 1.f : v);
          out[(size_t)grow * 3 + mi] = kf ? v * 255.f : 0.f;
        }
      }
    }
  }
}

extern "C" void kernel_launch(void* const* d_in, const int* in_sizes, int n_in,
                              void* d_out, int out_size, void* d_ws, size_t ws_size,
                              hipStream_t stream) {
  const float* fea  = (const float*)d_in[0];
  const float* pred = (const float*)d_in[1];
  const float* W1   = (const float*)d_in[2];
  const float* b1   = (const float*)d_in[3];
  const float* W2   = (const float*)d_in[4];
  const float* b2   = (const float*)d_in[5];
  const float* W3   = (const float*)d_in[6];
  const float* b3   = (const float*)d_in[7];
  const int*   tgt  = (const int*)d_in[8];
  const int*   pnum = (const int*)d_in[9];

  const int N     = in_sizes[1];       // 1048576
  const int n_per = N / 2;
  const int M     = in_sizes[8];       // N/2 scatter indices
  float* out = (float*)d_out;

  // workspace layout
  size_t NB = (size_t)N;
  unsigned char* ws    = (unsigned char*)d_ws;
  unsigned*      keys  = (unsigned*)ws;                        // 4N B
  unsigned char* flags = ws + 4 * NB;                          // N B
  unsigned char* keep  = ws + 5 * NB;                          // N B
  unsigned char* ktar  = ws + 6 * NB;                          // N B
  unsigned*      hist  = (unsigned*)(ws + 7 * NB);             // 2 KB
  unsigned*      state = (unsigned*)(ws + 7 * NB + 2048);      // 64 B
  float*         part  = (float*)(ws + 7 * NB + 4096);         // 8 KB
  half_t*        wbuf  = (half_t*)(ws + 7 * NB + 16384);       // 33792 B (halves)
  float*         bbuf  = (float*)(ws + 7 * NB + 16384 + 33792);// 704 B, contiguous with wbuf

  k_prep<<<72, 256, 0, stream>>>(W1, b1, W2, b2, W3, b3, wbuf, bbuf);
  k_localmax<<<(N / 8 + 255) / 256, 256, 0, stream>>>(pred, keys, flags, N / 8);
  k_zero_w<<<1024, 256, 0, stream>>>((unsigned*)ktar, N / 4);
  k_sel_init<<<1, 256, 0, stream>>>(hist, state, pnum, n_per);
  for (int pass = 0; pass < 4; pass++) {
    k_hist<<<1024, 256, 0, stream>>>(keys, state, hist, N, n_per, pass);
    k_select<<<1, 2, 0, stream>>>(hist, state, pass);
  }
  k_keep<<<(N + 255) / 256, 256, 0, stream>>>(pred, flags, state, keep, N, n_per);
  k_scatter<<<(M + 255) / 256, 256, 0, stream>>>(tgt, ktar, M);
  k_bce_partial<<<2048, 256, 0, stream>>>(pred, ktar, part, N);
  k_bce_final<<<1, 256, 0, stream>>>(part, out + (size_t)3 * N, 2048, N);
  k_mlp<<<N / MR, TPB, 0, stream>>>(fea, keep, wbuf, out, N);
}
